// Model_88459146428523
// MI455X (gfx1250) — matrix operand order repaired and checked by reading, not compiled, not run on hardware
//
#include <hip/hip_runtime.h>
#include <hip/hip_bf16.h>

typedef __attribute__((ext_vector_type(16))) _Float16 v16h;
typedef __attribute__((ext_vector_type(8)))  float    v8f;
typedef __attribute__((ext_vector_type(4)))  int      v4i_t;

// ---------------------------------------------------------------------------
// CDNA5 async global->LDS copy (ASYNCcnt path)
// ---------------------------------------------------------------------------
#if defined(__AMDGCN__) && defined(__has_builtin)
#if __has_builtin(__builtin_amdgcn_global_load_async_to_lds_b128) && \
    __has_builtin(__builtin_amdgcn_s_wait_asynccnt)
#define HAVE_ASYNC 1
#endif
#endif
#ifndef HAVE_ASYNC
#define HAVE_ASYNC 0
#endif

__device__ __forceinline__ void async_copy_b128(const void* g, void* lds) {
#if HAVE_ASYNC
  __builtin_amdgcn_global_load_async_to_lds_b128(
      (__attribute__((address_space(1))) v4i_t*)(unsigned long long)(g),
      (__attribute__((address_space(3))) v4i_t*)(unsigned int)(unsigned long long)(lds),
      0, 0);
#else
  *(uint4*)lds = *(const uint4*)g;
#endif
}
__device__ __forceinline__ void wait_async_le3() {
#if HAVE_ASYNC
  __builtin_amdgcn_s_wait_asynccnt(3);
#endif
}
__device__ __forceinline__ void wait_async_le0() {
#if HAVE_ASYNC
  __builtin_amdgcn_s_wait_asynccnt(0);
#endif
}

// ---------------------------------------------------------------------------
// WMMA fragment helpers (per CDNA5 ISA 7.12.2 VGPR layouts, wave32)
// ---------------------------------------------------------------------------
// A fragment: 16x32 f16, src row-major with leading dim ld (f16 elements).
__device__ __forceinline__ v16h load_a_frag(const _Float16* src, int ld, int lane) {
  int m = lane & 15, g = lane >> 4;
  const _Float16* p = src + m * ld;
  v16h a;
#pragma unroll
  for (int v = 0; v < 8; ++v) {
    int k = ((v < 4) ? 0 : 16) + g * 8 + (v & 3) * 2;
    a[2 * v]     = p[k];
    a[2 * v + 1] = p[k + 1];
  }
  return a;
}
// B fragment from pre-swizzled weights: tile (kt,nt) = 32 lanes x 16 f16,
// one contiguous 32-byte vector load per lane.
__device__ __forceinline__ v16h load_b_swz(const _Float16* wswz, int kt, int nt,
                                           int ntiles_n, int lane) {
  const v16h* p = (const v16h*)wswz;
  return p[(kt * ntiles_n + nt) * 32 + lane];
}
__device__ __forceinline__ v8f wmma_f16(v16h a, v16h b, v8f c) {
  return __builtin_amdgcn_wmma_f32_16x16x32_f16(false, a, false, b, (short)0, c,
                                                false, false);
}

__device__ __forceinline__ float angle3(float ax, float ay, float az,
                                        float bx, float by, float bz) {
  float cx = ay * bz - az * by;
  float cy = az * bx - ax * bz;
  float cz = ax * by - ay * bx;
  float cn = sqrtf(cx * cx + cy * cy + cz * cz);
  float d  = ax * bx + ay * by + az * bz;
  return atan2f(cn, d);
}

// ---------------------------------------------------------------------------
// Weight prep: f32 [sR,sC] -> f16 fragment-swizzled [Kp/32 x Np/16] tiles,
// each tile = 32 lanes x 16 contiguous f16 in B-fragment order.
// ---------------------------------------------------------------------------
__global__ void pad_convert_swz(const float* __restrict__ src, _Float16* __restrict__ dst,
                                int sR, int sC, int Kp, int Np) {
  int total = Kp * Np;
  int e = blockIdx.x * 256 + threadIdx.x;
  if (e >= total) return;
  int ntiles_n = Np >> 4;
  int el   = e & 15;
  int lane = (e >> 4) & 31;
  int tile = e >> 9;
  int nt = tile % ntiles_n, kt = tile / ntiles_n;
  int n = lane & 15, g = lane >> 4;
  int k = kt * 32 + ((el < 8) ? g * 8 + el : 16 + g * 8 + (el - 8));   // B element: table order, k = 8g+s / 16+8g+(s-8)
  int c = nt * 16 + n;
  float v = (k < sR && c < sC) ? src[k * sC + c] : 0.0f;
  dst[e] = (_Float16)v;
}

// ---------------------------------------------------------------------------
// lin_in: relu(relu(x@W0+b0)@W1+b1)  x:[N,16]f32 -> h0:[N,64]f16
// ---------------------------------------------------------------------------
__global__ void lin_in_kernel(const float* __restrict__ x,
                              const _Float16* __restrict__ w0s,  // swz K32 N64
                              const float* __restrict__ b0,
                              const _Float16* __restrict__ w1s,  // swz K64 N64
                              const float* __restrict__ b1,
                              _Float16* __restrict__ h0) {
  __shared__ _Float16 xa[8][16 * 32];
  __shared__ _Float16 h1[8][16 * 64];
  int wid = threadIdx.x >> 5, lane = threadIdx.x & 31;
  int row0 = blockIdx.x * 128 + wid * 16;

  for (int e = lane; e < 16 * 32; e += 32) {
    int r = e >> 5, c = e & 31;
    xa[wid][e] = (c < 16) ? (_Float16)x[(row0 + r) * 16 + c] : (_Float16)0.f;
  }
  __syncthreads();

  int n = lane & 15, g = lane >> 4;
  v16h a = load_a_frag(&xa[wid][0], 32, lane);
#pragma unroll
  for (int nt = 0; nt < 4; ++nt) {
    v16h b = load_b_swz(w0s, 0, nt, 4, lane);
    v8f acc = {};
    acc = wmma_f16(a, b, acc);
    float bias = b0[nt * 16 + n];
#pragma unroll
    for (int v = 0; v < 8; ++v)
      h1[wid][(v + 8 * g) * 64 + nt * 16 + n] = (_Float16)fmaxf(acc[v] + bias, 0.f);
  }
  __syncthreads();

  v16h a0 = load_a_frag(&h1[wid][0], 64, lane);
  v16h a1 = load_a_frag(&h1[wid][32], 64, lane);
#pragma unroll
  for (int nt = 0; nt < 4; ++nt) {
    v8f acc = {};
    acc = wmma_f16(a0, load_b_swz(w1s, 0, nt, 4, lane), acc);
    acc = wmma_f16(a1, load_b_swz(w1s, 1, nt, 4, lane), acc);
    float bias = b1[nt * 16 + n];
#pragma unroll
    for (int v = 0; v < 8; ++v)
      h0[(row0 + v + 8 * g) * 64 + nt * 16 + n] = (_Float16)fmaxf(acc[v] + bias, 0.f);
  }
}

// ---------------------------------------------------------------------------
// Farthest point sampling: single block, dists in dynamic LDS.
// ---------------------------------------------------------------------------
__global__ void fps_kernel(const float* __restrict__ pos, int n, int nsel,
                           int* __restrict__ sel) {
  extern __shared__ float dists[];  // n floats
  __shared__ float rv[32];
  __shared__ int   ri[32];
  __shared__ int   s_last;
  int tid = threadIdx.x, nt = blockDim.x;
  for (int j = tid; j < n; j += nt) dists[j] = 3.0e38f;
  if (tid == 0) { s_last = 0; sel[0] = 0; }
  __syncthreads();

  for (int it = 1; it < nsel; ++it) {
    int last = s_last;
    float lx = pos[last * 3 + 0], ly = pos[last * 3 + 1], lz = pos[last * 3 + 2];
    float bv = -1.0f; int bi = 0x7fffffff;
    for (int j = tid; j < n; j += nt) {
      float dx = pos[j * 3 + 0] - lx, dy = pos[j * 3 + 1] - ly, dz = pos[j * 3 + 2] - lz;
      float d = fminf(dists[j], dx * dx + dy * dy + dz * dz);
      dists[j] = d;
      if (d > bv) { bv = d; bi = j; }  // j ascending -> first max kept
    }
#pragma unroll
    for (int off = 16; off; off >>= 1) {
      float ov = __shfl_xor(bv, off, 32);
      int   oi = __shfl_xor(bi, off, 32);
      if (ov > bv || (ov == bv && oi < bi)) { bv = ov; bi = oi; }
    }
    int wid = tid >> 5, lane = tid & 31;
    if (lane == 0) { rv[wid] = bv; ri[wid] = bi; }
    __syncthreads();
    if (tid < 32) {
      int nw = nt >> 5;
      float v = (tid < nw) ? rv[tid] : -1.0f;
      int  ix = (tid < nw) ? ri[tid] : 0x7fffffff;
#pragma unroll
      for (int off = 16; off; off >>= 1) {
        float ov = __shfl_xor(v, off, 32);
        int   oi = __shfl_xor(ix, off, 32);
        if (ov > v || (ov == v && oi < ix)) { v = ov; ix = oi; }
      }
      if (tid == 0) { s_last = ix; sel[it] = ix; }
    }
    __syncthreads();
  }
}

__global__ void gather_kernel(const float* __restrict__ pos, const float* __restrict__ nrm,
                              const int* __restrict__ sel, int nq,
                              float* __restrict__ posq, float* __restrict__ nrmq) {
  int t = blockIdx.x * 256 + threadIdx.x;
  if (t >= nq) return;
  int s = sel[t];
#pragma unroll
  for (int k = 0; k < 3; ++k) {
    posq[t * 3 + k] = pos[s * 3 + k];
    nrmq[t * 3 + k] = nrm[s * 3 + k];
  }
}

// ---------------------------------------------------------------------------
// Radius top-32: one wave per query (8 queries/block).  Candidate positions
// streamed via double-buffered async global->LDS tiles (12KB each).
// ---------------------------------------------------------------------------
#define RTILE 1024
__global__ void radius_topk_kernel(const float* __restrict__ posq,
                                   const float* __restrict__ pos,
                                   int n, float r2,
                                   int* __restrict__ nbr,
                                   unsigned* __restrict__ vmask) {
  __shared__ float tile[2][RTILE * 3];
  int wid = threadIdx.x >> 5, lane = threadIdx.x & 31;
  int q = blockIdx.x * 8 + wid;
  float qx = posq[q * 3], qy = posq[q * 3 + 1], qz = posq[q * 3 + 2];

  float slot_d2 = 3.0e38f;
  int   slot_idx = 0;
  float curmax = 3.0e38f;
  int ntiles = n / RTILE;

  {
    const uint4* g = (const uint4*)pos;
    uint4* l = (uint4*)&tile[0][0];
#pragma unroll
    for (int r = 0; r < 3; ++r) {
      int s = threadIdx.x + r * 256;
      async_copy_b128(&g[s], &l[s]);
    }
  }

  for (int t = 0; t < ntiles; ++t) {
    if (t + 1 < ntiles) {
      const uint4* g = (const uint4*)(pos + (size_t)(t + 1) * RTILE * 3);
      uint4* l = (uint4*)&tile[(t + 1) & 1][0];
#pragma unroll
      for (int r = 0; r < 3; ++r) {
        int s = threadIdx.x + r * 256;
        async_copy_b128(&g[s], &l[s]);
      }
      wait_async_le3();
    } else {
      wait_async_le0();
    }
    __syncthreads();

    const float* tp = &tile[t & 1][0];
    int base = t * RTILE;
    for (int c0 = 0; c0 < RTILE; c0 += 32) {
      int cl = c0 + lane;
      float dx = tp[cl * 3]     - qx;
      float dy = tp[cl * 3 + 1] - qy;
      float dz = tp[cl * 3 + 2] - qz;
      float d2 = dx * dx + dy * dy + dz * dz;
      int cand = base + cl;
      unsigned m = (unsigned)__ballot(d2 < curmax);
      while (m) {
        int src = __ffs(m) - 1;
        m &= m - 1;
        float cd2 = __shfl(d2, src, 32);
        int   ci  = __shfl(cand, src, 32);
        if (cd2 < curmax) {
          float mv = slot_d2; int ml = lane;
#pragma unroll
          for (int off = 16; off; off >>= 1) {
            float ov = __shfl_xor(mv, off, 32);
            int   ol = __shfl_xor(ml, off, 32);
            if (ov > mv || (ov == mv && ol > ml)) { mv = ov; ml = ol; }
          }
          if (lane == ml) { slot_d2 = cd2; slot_idx = ci; }
          float nm = slot_d2;
#pragma unroll
          for (int off = 16; off; off >>= 1) nm = fmaxf(nm, __shfl_xor(nm, off, 32));
          curmax = nm;
        }
      }
    }
    __syncthreads();
  }

  nbr[q * 32 + lane] = slot_idx;
  unsigned vm = (unsigned)__ballot(slot_d2 <= r2);
  if (lane == 0) vmask[q] = vm;
}

// ---------------------------------------------------------------------------
// PPF message build + nn1 (2x 68x68) + masked max-agg + nn2 (68x64), all WMMA.
// 4 queries/block -> 128 message rows, K padded to 96, N padded to 80.
// ---------------------------------------------------------------------------
#define MQB 4
__global__ void msg_nn_kernel(const _Float16* __restrict__ xCur,
                              const float* __restrict__ posCur,
                              const float* __restrict__ normCur,
                              const float* __restrict__ posQ,
                              const float* __restrict__ normQ,
                              const int* __restrict__ nbr,
                              const unsigned* __restrict__ vmask,
                              const _Float16* __restrict__ w1a, const float* __restrict__ b1a,
                              const _Float16* __restrict__ w1b, const float* __restrict__ b1b,
                              const _Float16* __restrict__ w2,  const float* __restrict__ b2,
                              _Float16* __restrict__ xNext) {
  __shared__ _Float16 bufA[128 * 96];
  __shared__ _Float16 bufB[128 * 96];
  __shared__ _Float16 agg[16 * 96];
  __shared__ float    qp[MQB][3], qn[MQB][3];
  __shared__ unsigned vm_s[MQB];

  int tid = threadIdx.x;
  int q0 = blockIdx.x * MQB;
  if (tid < MQB) {
    vm_s[tid] = vmask[q0 + tid];
#pragma unroll
    for (int k = 0; k < 3; ++k) {
      qp[tid][k] = posQ[(q0 + tid) * 3 + k];
      qn[tid][k] = normQ[(q0 + tid) * 3 + k];
    }
  }
  __syncthreads();

  // ---- build message rows (2 threads per row) ----
  {
    int r = tid >> 1, half = tid & 1;
    int q = r >> 5, ks = r & 31;
    int j = nbr[(q0 + q) * 32 + ks];
    _Float16* row = &bufA[r * 96];
    const uint2* src = (const uint2*)(xCur + (size_t)j * 64);
    uint2* drow = (uint2*)row;
#pragma unroll
    for (int u = 0; u < 8; ++u) drow[half * 8 + u] = src[half * 8 + u];
    if (half == 0) {
      float px = posCur[j * 3], py = posCur[j * 3 + 1], pz = posCur[j * 3 + 2];
      float nx = normCur[j * 3], ny = normCur[j * 3 + 1], nz = normCur[j * 3 + 2];
      float dx = px - qp[q][0], dy = py - qp[q][1], dz = pz - qp[q][2];
      float ax = qn[q][0], ay = qn[q][1], az = qn[q][2];
      row[64] = (_Float16)sqrtf(dx * dx + dy * dy + dz * dz);
      row[65] = (_Float16)angle3(ax, ay, az, dx, dy, dz);
      row[66] = (_Float16)angle3(nx, ny, nz, dx, dy, dz);
      row[67] = (_Float16)angle3(ax, ay, az, nx, ny, nz);
    } else {
      for (int c = 68; c < 96; ++c) row[c] = (_Float16)0.f;
      _Float16* rowB = &bufB[r * 96];
      for (int c = 80; c < 96; ++c) rowB[c] = (_Float16)0.f;
    }
  }
  __syncthreads();

  int wid = tid >> 5, lane = tid & 31;
  int n16 = lane & 15, g = lane >> 4;

  // ---- nn1 layer 1: [128,96] x [96,80] -> bufB ----
  {
    int mt = wid;  // 8 waves, 8 mtiles
    v16h a0 = load_a_frag(&bufA[(mt * 16) * 96 +  0], 96, lane);
    v16h a1 = load_a_frag(&bufA[(mt * 16) * 96 + 32], 96, lane);
    v16h a2 = load_a_frag(&bufA[(mt * 16) * 96 + 64], 96, lane);
#pragma unroll
    for (int nt = 0; nt < 5; ++nt) {
      v8f acc = {};
      acc = wmma_f16(a0, load_b_swz(w1a, 0, nt, 5, lane), acc);
      acc = wmma_f16(a1, load_b_swz(w1a, 1, nt, 5, lane), acc);
      acc = wmma_f16(a2, load_b_swz(w1a, 2, nt, 5, lane), acc);
      int col = nt * 16 + n16;
      float bias = (col < 68) ? b1a[col] : 0.f;
#pragma unroll
      for (int v = 0; v < 8; ++v)
        bufB[(mt * 16 + v + 8 * g) * 96 + col] = (_Float16)fmaxf(acc[v] + bias, 0.f);
    }
  }
  __syncthreads();

  // ---- nn1 layer 2: bufB x w1b -> bufA ----
  {
    int mt = wid;
    v16h a0 = load_a_frag(&bufB[(mt * 16) * 96 +  0], 96, lane);
    v16h a1 = load_a_frag(&bufB[(mt * 16) * 96 + 32], 96, lane);
    v16h a2 = load_a_frag(&bufB[(mt * 16) * 96 + 64], 96, lane);
#pragma unroll
    for (int nt = 0; nt < 5; ++nt) {
      v8f acc = {};
      acc = wmma_f16(a0, load_b_swz(w1b, 0, nt, 5, lane), acc);
      acc = wmma_f16(a1, load_b_swz(w1b, 1, nt, 5, lane), acc);
      acc = wmma_f16(a2, load_b_swz(w1b, 2, nt, 5, lane), acc);
      int col = nt * 16 + n16;
      float bias = (col < 68) ? b1b[col] : 0.f;
#pragma unroll
      for (int v = 0; v < 8; ++v)
        bufA[(mt * 16 + v + 8 * g) * 96 + col] = (_Float16)fmaxf(acc[v] + bias, 0.f);
    }
  }
  __syncthreads();

  // ---- masked max aggregation over 32 neighbors -> agg [16][96] ----
  for (int e = tid; e < 16 * 96; e += 256) {
    int r = e / 96, c = e % 96;
    float v = 0.f;
    if (r < MQB && c < 80) {
      unsigned m = vm_s[r];
      float mx = 0.f;
      while (m) {
        int k = __ffs(m) - 1;
        m &= m - 1;
        mx = fmaxf(mx, (float)bufA[(r * 32 + k) * 96 + c]);
      }
      v = mx;  // relu'd values are >=0; empty -> 0 (matches reference)
    }
    agg[e] = (_Float16)v;
  }
  __syncthreads();

  // ---- nn2: [16,96] x [96,64] -> xNext (rows 0..MQB-1 valid) ----
  if (wid < 4) {
    int nt = wid;
    v16h a0 = load_a_frag(&agg[0],  96, lane);
    v16h a1 = load_a_frag(&agg[32], 96, lane);
    v16h a2 = load_a_frag(&agg[64], 96, lane);
    v8f acc = {};
    acc = wmma_f16(a0, load_b_swz(w2, 0, nt, 4, lane), acc);
    acc = wmma_f16(a1, load_b_swz(w2, 1, nt, 4, lane), acc);
    acc = wmma_f16(a2, load_b_swz(w2, 2, nt, 4, lane), acc);
    int col = nt * 16 + n16;
    float bias = b2[col];
    if (g == 0) {
#pragma unroll
      for (int v = 0; v < MQB; ++v)
        xNext[(size_t)(q0 + v) * 64 + col] = (_Float16)fmaxf(acc[v] + bias, 0.f);
    }
  }
}

// ---------------------------------------------------------------------------
// mean-pool over [4096,64] + lin_out head -> out[2]
// ---------------------------------------------------------------------------
__global__ void final_kernel(const _Float16* __restrict__ x3,
                             const float* __restrict__ w0, const float* __restrict__ b0,
                             const float* __restrict__ w1, const float* __restrict__ b1,
                             float* __restrict__ out) {
  __shared__ float part[4][64];
  __shared__ float mean_s[64];
  __shared__ float h_s[64];
  int t = threadIdx.x, col = t & 63, grp = t >> 6;
  float s = 0.f;
  for (int r = grp; r < 4096; r += 4) s += (float)x3[r * 64 + col];
  part[grp][col] = s;
  __syncthreads();
  if (t < 64)
    mean_s[t] = (part[0][t] + part[1][t] + part[2][t] + part[3][t]) * (1.0f / 4096.0f);
  __syncthreads();
  if (t < 64) {
    float a = b0[t];
    for (int k = 0; k < 64; ++k) a += mean_s[k] * w0[k * 64 + t];
    h_s[t] = fmaxf(a, 0.f);
  }
  __syncthreads();
  if (t < 2) {
    float a = b1[t];
    for (int k = 0; k < 64; ++k) a += h_s[k] * w1[k * 2 + t];
    out[t] = a;
  }
}

// ---------------------------------------------------------------------------
// Host side
// ---------------------------------------------------------------------------
extern "C" void kernel_launch(void* const* d_in, const int* in_sizes, int n_in,
                              void* d_out, int out_size, void* d_ws, size_t ws_size,
                              hipStream_t stream) {
  (void)in_sizes; (void)n_in; (void)out_size; (void)ws_size;
  const float* x       = (const float*)d_in[0];
  const float* pos     = (const float*)d_in[1];
  const float* norm    = (const float*)d_in[2];
  const float* li_w0   = (const float*)d_in[3];
  const float* li_b0   = (const float*)d_in[4];
  const float* li_w1   = (const float*)d_in[5];
  const float* li_b1   = (const float*)d_in[6];
  const float* nn1_w0  = (const float*)d_in[7];   // [3][68][68]
  const float* nn1_b0  = (const float*)d_in[8];   // [3][68]
  const float* nn1_w1  = (const float*)d_in[9];
  const float* nn1_b1  = (const float*)d_in[10];
  const float* nn2_w   = (const float*)d_in[11];  // [3][68][64]
  const float* nn2_b   = (const float*)d_in[12];  // [3][64]
  const float* lo_w0   = (const float*)d_in[13];
  const float* lo_b0   = (const float*)d_in[14];
  const float* lo_w1   = (const float*)d_in[15];
  const float* lo_b1   = (const float*)d_in[16];
  float* out = (float*)d_out;

  const int N = 32768;
  const int ns[4] = {32768, 16384, 8192, 4096};
  const float r2s[3] = {4.0f, 16.0f, 64.0f};

  char* ws = (char*)d_ws;
  auto alloc = [&](size_t bytes) -> char* {
    char* p = ws;
    ws += (bytes + 255) & ~(size_t)255;
    return p;
  };

  _Float16* xL[4];
  xL[0] = (_Float16*)alloc((size_t)ns[0] * 64 * 2);
  xL[1] = (_Float16*)alloc((size_t)ns[1] * 64 * 2);
  xL[2] = (_Float16*)alloc((size_t)ns[2] * 64 * 2);
  xL[3] = (_Float16*)alloc((size_t)ns[3] * 64 * 2);
  float* posL[4]; float* normL[4];
  posL[0]  = (float*)pos;  normL[0] = (float*)norm;
  for (int i = 1; i < 4; ++i) {
    posL[i]  = (float*)alloc((size_t)ns[i] * 3 * 4);
    normL[i] = (float*)alloc((size_t)ns[i] * 3 * 4);
  }
  int* sel[3]; int* nbrL[3]; unsigned* vmL[3];
  for (int i = 0; i < 3; ++i) {
    sel[i]  = (int*)alloc((size_t)ns[i + 1] * 4);
    nbrL[i] = (int*)alloc((size_t)ns[i + 1] * 32 * 4);
    vmL[i]  = (unsigned*)alloc((size_t)ns[i + 1] * 4);
  }
  _Float16* wi0s = (_Float16*)alloc(32 * 64 * 2);
  _Float16* wi1s = (_Float16*)alloc(64 * 64 * 2);
  _Float16* w1as[3]; _Float16* w1bs[3]; _Float16* w2s[3];
  for (int i = 0; i < 3; ++i) {
    w1as[i] = (_Float16*)alloc(96 * 80 * 2);
    w1bs[i] = (_Float16*)alloc(96 * 80 * 2);
    w2s[i]  = (_Float16*)alloc(96 * 64 * 2);
  }

  // --- weight prep (fragment-swizzled) ---
  pad_convert_swz<<<(32 * 64 + 255) / 256, 256, 0, stream>>>(li_w0, wi0s, 16, 64, 32, 64);
  pad_convert_swz<<<(64 * 64 + 255) / 256, 256, 0, stream>>>(li_w1, wi1s, 64, 64, 64, 64);
  for (int i = 0; i < 3; ++i) {
    pad_convert_swz<<<(96 * 80 + 255) / 256, 256, 0, stream>>>(
        nn1_w0 + (size_t)i * 68 * 68, w1as[i], 68, 68, 96, 80);
    pad_convert_swz<<<(96 * 80 + 255) / 256, 256, 0, stream>>>(
        nn1_w1 + (size_t)i * 68 * 68, w1bs[i], 68, 68, 96, 80);
    pad_convert_swz<<<(96 * 64 + 255) / 256, 256, 0, stream>>>(
        nn2_w + (size_t)i * 68 * 64, w2s[i], 68, 64, 96, 64);
  }

  // --- lin_in ---
  lin_in_kernel<<<N / 128, 256, 0, stream>>>(x, wi0s, li_b0, wi1s, li_b1, xL[0]);

  // --- SA levels ---
  for (int i = 0; i < 3; ++i) {
    int n = ns[i], nq = ns[i + 1];
    fps_kernel<<<1, 1024, (size_t)n * 4, stream>>>(posL[i], n, nq, sel[i]);
    gather_kernel<<<(nq + 255) / 256, 256, 0, stream>>>(posL[i], normL[i], sel[i], nq,
                                                        posL[i + 1], normL[i + 1]);
    radius_topk_kernel<<<nq / 8, 256, 0, stream>>>(posL[i + 1], posL[i], n, r2s[i],
                                                   nbrL[i], vmL[i]);
    msg_nn_kernel<<<nq / MQB, 256, 0, stream>>>(
        xL[i], posL[i], normL[i], posL[i + 1], normL[i + 1], nbrL[i], vmL[i],
        w1as[i], nn1_b0 + (size_t)i * 68, w1bs[i], nn1_b1 + (size_t)i * 68,
        w2s[i], nn2_b + (size_t)i * 64, xL[i + 1]);
  }

  // --- pool + head ---
  final_kernel<<<1, 256, 0, stream>>>(xL[3], lo_w0, lo_b0, lo_w1, lo_b1, out);
}
